// TransformerDecoderBlock_54288386621958
// MI455X (gfx1250) — matrix operand order repaired and checked by reading, not compiled, not run on hardware
//
#include <hip/hip_runtime.h>
#include <hip/hip_bf16.h>
#include <math.h>

#define B_      32
#define DMODEL  2048
#define H_      16
#define KVH_    4
#define HD_     128
#define NREP_   4
#define DFF_    8192
#define LCACHE_ 4096
#define EPS_    1e-6f

typedef __attribute__((ext_vector_type(16))) __bf16 v16bf;
typedef __attribute__((ext_vector_type(8)))  float  v8f;
typedef unsigned int u32;
typedef __attribute__((ext_vector_type(4))) u32 u32x4;
typedef __attribute__((ext_vector_type(8))) int i32x8;
typedef __attribute__((ext_vector_type(4))) int i32x4;

#if defined(__gfx1250__) && __has_builtin(__builtin_amdgcn_tensor_load_to_lds)
#define HAVE_TDM 1
#else
#define HAVE_TDM 0
#endif

// Convert 8 contiguous f32 -> bf16 into elements [O, O+8) of a v16bf.
template <int O>
__device__ __forceinline__ void cvt8(v16bf& d, const float* __restrict__ p) {
  float4 x = ((const float4*)p)[0];
  float4 y = ((const float4*)p)[1];
  d[O + 0] = (__bf16)x.x; d[O + 1] = (__bf16)x.y;
  d[O + 2] = (__bf16)x.z; d[O + 3] = (__bf16)x.w;
  d[O + 4] = (__bf16)y.x; d[O + 5] = (__bf16)y.y;
  d[O + 6] = (__bf16)y.z; d[O + 7] = (__bf16)y.w;
}

#if HAVE_TDM
__device__ __forceinline__ u32 rfl(u32 x) {
  return (u32)__builtin_amdgcn_readfirstlane((int)x);
}
// Issue a TDM DMA of a [rows x cols] f32 tile (row stride = ld floats) from
// global memory into LDS at byte offset lds_off, packed contiguously.
// D# bitfields per CDNA5 ISA sec. 8.3/8.4 (count=1, type=2, data_size=4B).
// All descriptor dwords are forced wave-uniform (SGPR D#) via readfirstlane.
__device__ __forceinline__ void tdm_load_tile_f32(const float* gsrc, u32 lds_off,
                                                  u32 ld, u32 rows, u32 cols) {
  unsigned long long ga = (unsigned long long)(__UINTPTR_TYPE__)gsrc;
  u32 ga_lo = rfl((u32)(ga & 0xFFFFFFFFull));
  u32 ga_hi = rfl((u32)(ga >> 32));
  lds_off   = rfl(lds_off);
  ld        = rfl(ld);
  u32x4 g0;
  g0[0] = 1u;                                            // count=1, user mode
  g0[1] = lds_off;                                       // lds_addr
  g0[2] = ga_lo;                                         // global_addr[31:0]
  g0[3] = (ga_hi & 0x1FFFFFFu) | (2u << 30);             // addr[56:32] | type=2
  i32x8 g1;
  g1[0] = (int)(2u << 16);                               // wg_mask=0, data_size=4B
  g1[1] = (int)((ld & 0xFFFFu) << 16);                   // tensor_dim0[15:0]
  g1[2] = (int)(((ld >> 16) & 0xFFFFu) | ((rows & 0xFFFFu) << 16)); // dim0 hi|dim1 lo
  g1[3] = (int)((cols & 0xFFFFu) << 16);                 // dim1 hi(0) | tile_dim0
  g1[4] = (int)(rows & 0xFFFFu);                         // tile_dim1 | tile_dim2=0
  g1[5] = (int)ld;                                       // tensor_dim0_stride lo
  g1[6] = 0;                                             // stride0 hi | stride1 lo
  g1[7] = 0;                                             // stride1 hi
  i32x4 gz = {0, 0, 0, 0};
#if __clang_major__ >= 23
  i32x8 gz8 = {0, 0, 0, 0, 0, 0, 0, 0};
  __builtin_amdgcn_tensor_load_to_lds(g0, g1, gz, gz, gz8, 0);
#else
  __builtin_amdgcn_tensor_load_to_lds(g0, g1, gz, gz, 0);
#endif
}
#endif

// ---------------------------------------------------------------------------
// RMSNorm: out[b,:] = x[b,:] * rsqrt(mean(x^2)+eps) * w
// ---------------------------------------------------------------------------
__global__ void rmsnorm_kernel(float* __restrict__ out, const float* __restrict__ x,
                               const float* __restrict__ w) {
  __shared__ float s_red[256];
  const int b = blockIdx.x, tid = threadIdx.x;
  const float* xr = x + (size_t)b * DMODEL;
  float ss = 0.f;
  for (int i = tid; i < DMODEL; i += 256) { float v = xr[i]; ss += v * v; }
  s_red[tid] = ss; __syncthreads();
  for (int s = 128; s > 0; s >>= 1) {
    if (tid < s) s_red[tid] += s_red[tid + s];
    __syncthreads();
  }
  const float rs = rsqrtf(s_red[0] * (1.0f / DMODEL) + EPS_);
  float* o = out + (size_t)b * DMODEL;
  for (int i = tid; i < DMODEL; i += 256) o[i] = xr[i] * rs * w[i];
}

// ---------------------------------------------------------------------------
// WMMA GEMM: C(32 x N) = A(32 x K) @ W(N x K)^T [+ bias[N]] [+ resid(32 x N)]
// 128-thread block = 4 waves, 4 adjacent 16-wide N tiles.
// Uniform TDM pipeline: every wave DMAs one 4-row slice of the shared 16x32 A
// tile plus its own 16x32 W tile, double-buffered, so each wave issues exactly
// 2 TDM ops per K step and executes s_wait_tensorcnt(2) (wait(0) on the last step)
// (in-order TDM completion -> the current pair is done, next pair in flight).
// ---------------------------------------------------------------------------
template <bool HAS_BIAS, bool HAS_RESID>
__global__ void wmma_gemm_kernel(float* __restrict__ C, const float* __restrict__ A,
                                 const float* __restrict__ W,
                                 const float* __restrict__ bias,
                                 const float* __restrict__ resid, int N, int K) {
  const int tid    = threadIdx.x;      // 0..127
  const int lane   = tid & 31;
  const int wid    = tid >> 5;         // 0..3: N-tile within block
  const int tile_m = blockIdx.y;
  const int nl     = lane & 15;
  const int half   = lane >> 4;
  const int col0   = (blockIdx.x * 4 + wid) * 16;
  const int col    = col0 + nl;
  const float* Atile = A + (size_t)(tile_m * 16) * K;
  const float* Wtile = W + (size_t)col0 * K;

  v8f acc = {0.f, 0.f, 0.f, 0.f, 0.f, 0.f, 0.f, 0.f};
#if HAVE_TDM
  __shared__ float s_a[2][16 * 32];
  __shared__ float s_w[4][2][16 * 32];
  const int nit = K / 32;
  const float* Aq = Atile + (size_t)(wid * 4) * K;     // this wave's 4-row slice
  const u32 a_off[2] = {(u32)(__UINTPTR_TYPE__)&s_a[0][wid * 4 * 32],
                        (u32)(__UINTPTR_TYPE__)&s_a[1][wid * 4 * 32]};
  const u32 w_off[2] = {(u32)(__UINTPTR_TYPE__)&s_w[wid][0][0],
                        (u32)(__UINTPTR_TYPE__)&s_w[wid][1][0]};
  tdm_load_tile_f32(Aq,    a_off[0], (u32)K, 4u, 32u);
  tdm_load_tile_f32(Wtile, w_off[0], (u32)K, 16u, 32u);
  for (int it = 0; it < nit; ++it) {
    const int cur = it & 1;
    if (it + 1 < nit) {
      const int k1 = (it + 1) * 32;
      tdm_load_tile_f32(Aq + k1,    a_off[cur ^ 1], (u32)K, 4u, 32u);
      tdm_load_tile_f32(Wtile + k1, w_off[cur ^ 1], (u32)K, 16u, 32u);
      __builtin_amdgcn_s_wait_tensorcnt(2);   // current pair done, next in flight
    } else {
      __builtin_amdgcn_s_wait_tensorcnt(0);
    }
    __syncthreads();                      // all A slices published
    v16bf a, b;
    // A 16x32 bf16 layout: lanes 0-15 K={0..15}, lanes 16-31 K={16..31}
    const float* arow = &s_a[cur][nl * 32];
    cvt8<0>(a, arow + 16 * half);
    cvt8<8>(a, arow + 16 * half + 8);
    // B 32x16 bf16 layout: lane n holds 16 contiguous K for its column
    const float* wrow = &s_w[wid][cur][nl * 32 + 16 * half];
    cvt8<0>(b, wrow);
    cvt8<8>(b, wrow + 8);
    acc = __builtin_amdgcn_wmma_f32_16x16x32_bf16(false, a, false, b,
                                                  (short)0, acc, false, false);
    __syncthreads();                      // A_cur fully consumed before re-DMA
  }
#else
  __shared__ float s_a1[16 * 32];
  const float* Wrow = Wtile + (size_t)nl * K + 16 * half;
  for (int k0 = 0; k0 < K; k0 += 32) {
    __syncthreads();
    {
      const int r = tid >> 3, sg = (tid & 7) * 4;
      *(float4*)&s_a1[r * 32 + sg] = *(const float4*)(Atile + (size_t)r * K + k0 + sg);
    }
    __syncthreads();
    __builtin_prefetch(Wrow + k0 + 128, 0, 1);
    v16bf a, b;
    const float* arow = &s_a1[nl * 32];
    cvt8<0>(a, arow + 16 * half);
    cvt8<8>(a, arow + 16 * half + 8);
    cvt8<0>(b, Wrow + k0);
    cvt8<8>(b, Wrow + k0 + 8);
    acc = __builtin_amdgcn_wmma_f32_16x16x32_bf16(false, a, false, b,
                                                  (short)0, acc, false, false);
  }
#endif
  const float bv = HAS_BIAS ? bias[col] : 0.f;
#pragma unroll
  for (int r = 0; r < 8; ++r) {
    const int row = tile_m * 16 + 8 * half + r;          // C/D layout: VGPR r -> M
    float v = acc[r] + bv;
    if (HAS_RESID) v += resid[(size_t)row * N + col];
    C[(size_t)row * N + col] = v;
  }
}

// ---------------------------------------------------------------------------
// Fused gate/up WMMA GEMM + SiLU: act = (f@gw^T+gb) * silu(f@uw^T+ub)
// Same uniform TDM pipeline: per wave (A-slice + Wg + Wu) per step, wait(3).
// ---------------------------------------------------------------------------
__global__ void wmma_gateup_kernel(float* __restrict__ act, const float* __restrict__ A,
                                   const float* __restrict__ gw, const float* __restrict__ gb,
                                   const float* __restrict__ uw, const float* __restrict__ ub) {
  const int K      = DMODEL;
  const int tid    = threadIdx.x;
  const int lane   = tid & 31;
  const int wid    = tid >> 5;
  const int tile_m = blockIdx.y;
  const int nl     = lane & 15;
  const int half   = lane >> 4;
  const int col0   = (blockIdx.x * 4 + wid) * 16;
  const int col    = col0 + nl;
  const float* Atile = A + (size_t)(tile_m * 16) * K;
  const float* Gtile = gw + (size_t)col0 * K;
  const float* Utile = uw + (size_t)col0 * K;

  v8f accg = {0.f, 0.f, 0.f, 0.f, 0.f, 0.f, 0.f, 0.f};
  v8f accu = {0.f, 0.f, 0.f, 0.f, 0.f, 0.f, 0.f, 0.f};
#if HAVE_TDM
  __shared__ float s_a[2][16 * 32];
  __shared__ float s_wg[4][2][16 * 32];
  __shared__ float s_wu[4][2][16 * 32];
  const int nit = K / 32;
  const float* Aq = Atile + (size_t)(wid * 4) * K;
  const u32 a_off[2] = {(u32)(__UINTPTR_TYPE__)&s_a[0][wid * 4 * 32],
                        (u32)(__UINTPTR_TYPE__)&s_a[1][wid * 4 * 32]};
  const u32 g_off[2] = {(u32)(__UINTPTR_TYPE__)&s_wg[wid][0][0],
                        (u32)(__UINTPTR_TYPE__)&s_wg[wid][1][0]};
  const u32 u_off[2] = {(u32)(__UINTPTR_TYPE__)&s_wu[wid][0][0],
                        (u32)(__UINTPTR_TYPE__)&s_wu[wid][1][0]};
  tdm_load_tile_f32(Aq,    a_off[0], (u32)K, 4u, 32u);
  tdm_load_tile_f32(Gtile, g_off[0], (u32)K, 16u, 32u);
  tdm_load_tile_f32(Utile, u_off[0], (u32)K, 16u, 32u);
  for (int it = 0; it < nit; ++it) {
    const int cur = it & 1;
    if (it + 1 < nit) {
      const int k1 = (it + 1) * 32;
      tdm_load_tile_f32(Aq + k1,    a_off[cur ^ 1], (u32)K, 4u, 32u);
      tdm_load_tile_f32(Gtile + k1, g_off[cur ^ 1], (u32)K, 16u, 32u);
      tdm_load_tile_f32(Utile + k1, u_off[cur ^ 1], (u32)K, 16u, 32u);
      __builtin_amdgcn_s_wait_tensorcnt(3);   // current triple done
    } else {
      __builtin_amdgcn_s_wait_tensorcnt(0);
    }
    __syncthreads();
    v16bf a, bg, bu;
    const float* arow = &s_a[cur][nl * 32];
    cvt8<0>(a, arow + 16 * half);
    cvt8<8>(a, arow + 16 * half + 8);
    const float* grow = &s_wg[wid][cur][nl * 32 + 16 * half];
    const float* urow = &s_wu[wid][cur][nl * 32 + 16 * half];
    cvt8<0>(bg, grow); cvt8<8>(bg, grow + 8);
    cvt8<0>(bu, urow); cvt8<8>(bu, urow + 8);
    accg = __builtin_amdgcn_wmma_f32_16x16x32_bf16(false, a, false, bg,
                                                   (short)0, accg, false, false);
    accu = __builtin_amdgcn_wmma_f32_16x16x32_bf16(false, a, false, bu,
                                                   (short)0, accu, false, false);
    __syncthreads();
  }
#else
  __shared__ float s_a1[16 * 32];
  const float* Grow = Gtile + (size_t)nl * K + 16 * half;
  const float* Urow = Utile + (size_t)nl * K + 16 * half;
  for (int k0 = 0; k0 < K; k0 += 32) {
    __syncthreads();
    {
      const int r = tid >> 3, sg = (tid & 7) * 4;
      *(float4*)&s_a1[r * 32 + sg] = *(const float4*)(Atile + (size_t)r * K + k0 + sg);
    }
    __syncthreads();
    __builtin_prefetch(Grow + k0 + 128, 0, 1);
    __builtin_prefetch(Urow + k0 + 128, 0, 1);
    v16bf a, bg, bu;
    const float* arow = &s_a1[nl * 32];
    cvt8<0>(a, arow + 16 * half);
    cvt8<8>(a, arow + 16 * half + 8);
    cvt8<0>(bg, Grow + k0); cvt8<8>(bg, Grow + k0 + 8);
    cvt8<0>(bu, Urow + k0); cvt8<8>(bu, Urow + k0 + 8);
    accg = __builtin_amdgcn_wmma_f32_16x16x32_bf16(false, a, false, bg,
                                                   (short)0, accg, false, false);
    accu = __builtin_amdgcn_wmma_f32_16x16x32_bf16(false, a, false, bu,
                                                   (short)0, accu, false, false);
  }
#endif
  const float biasg = gb[col], biasu = ub[col];
#pragma unroll
  for (int r = 0; r < 8; ++r) {
    const int row = tile_m * 16 + 8 * half + r;
    const float g = accg[r] + biasg;
    const float u = accu[r] + biasu;
    const float sil = u / (1.f + __expf(-u));
    act[(size_t)row * DFF_ + col] = g * sil;
  }
}

// ---------------------------------------------------------------------------
// RoPE + per-head RMSNorm for q (16 heads) and k (4 heads), in place.
// ---------------------------------------------------------------------------
__global__ void rope_norm_kernel(float* __restrict__ qbuf, float* __restrict__ kbuf,
                                 const float* __restrict__ cos_t,
                                 const float* __restrict__ sin_t) {
  __shared__ float s_t[HD_];
  __shared__ float s_red[HD_];
  const int idx = blockIdx.x;
  const int b   = idx / (H_ + KVH_);
  const int hh  = idx % (H_ + KVH_);
  const int d   = threadIdx.x;

  float* buf; int base; int hc;
  if (hh < H_) { buf = qbuf; base = b * DMODEL + hh * HD_;          hc = hh; }
  else { int h = hh - H_; buf = kbuf; base = b * (KVH_ * HD_) + h * HD_; hc = h; }

  s_t[d] = buf[base + d];
  __syncthreads();

  const int dd = (d < HD_ / 2) ? d : d - HD_ / 2;
  const float c = cos_t[(size_t)b * H_ * HD_ + hc * HD_ + dd];
  const float s = sin_t[(size_t)b * H_ * HD_ + hc * HD_ + dd];
  float r;
  if (d < HD_ / 2) r = s_t[d] * c - s_t[d + HD_ / 2] * s;
  else             r = s_t[d] * c + s_t[d - HD_ / 2] * s;

  s_red[d] = r * r;
  __syncthreads();
  for (int st = HD_ / 2; st > 0; st >>= 1) {
    if (d < st) s_red[d] += s_red[d + st];
    __syncthreads();
  }
  const float scale = rsqrtf(s_red[0] * (1.0f / HD_) + EPS_);
  buf[base + d] = r * scale;
}

// ---------------------------------------------------------------------------
// Attention: one block per (head, batch). Scores pass: one wave per K row
// (32 lanes x float4 = the whole 128-float row, fully coalesced), dot reduced
// with a __shfl_xor butterfly. Fresh k/v row substituted at j == pos.
// ---------------------------------------------------------------------------
__global__ void attn_kernel(float* __restrict__ out, const float* __restrict__ qbuf,
                            const float* __restrict__ kbuf, const float* __restrict__ vbuf,
                            const float* __restrict__ k_cache,
                            const float* __restrict__ v_cache,
                            const int* __restrict__ pos) {
  __shared__ float s_scores[LCACHE_];
  __shared__ float s_red[256];
  __shared__ float s_part[2][HD_];

  const int h = blockIdx.x, b = blockIdx.y, tid = threadIdx.x;
  const int lane = tid & 31, w = tid >> 5;   // 8 waves
  const int p   = pos[b];
  const int kvh = h / NREP_;
  const float inv_sqrt_hd = rsqrtf((float)HD_);

  // Per-lane q fragment: dims lane*4 .. lane*4+3
  const float* qrow = qbuf + (size_t)b * DMODEL + h * HD_;
  const float4 qv = *(const float4*)(qrow + lane * 4);

  const float* kc   = k_cache + ((size_t)b * KVH_ + kvh) * LCACHE_ * HD_;
  const float* knew = kbuf + (size_t)b * (KVH_ * HD_) + kvh * HD_;

  float lmax = -INFINITY;
  for (int j = w; j <= p; j += 8) {
    const float* kr = (j == p) ? knew : (kc + (size_t)j * HD_);
    const float4 kk = *(const float4*)(kr + lane * 4);
    float part = kk.x * qv.x + kk.y * qv.y + kk.z * qv.z + kk.w * qv.w;
#pragma unroll
    for (int m = 16; m > 0; m >>= 1) part += __shfl_xor(part, m, 32);
    const float sc = part * inv_sqrt_hd;
    if (lane == 0) s_scores[j] = sc;
    lmax = fmaxf(lmax, sc);
  }
  s_red[tid] = lmax;
  __syncthreads();
  for (int s = 128; s > 0; s >>= 1) {
    if (tid < s) s_red[tid] = fmaxf(s_red[tid], s_red[tid + s]);
    __syncthreads();
  }
  const float m = s_red[0];
  __syncthreads();

  float lsum = 0.f;
  for (int j = tid; j <= p; j += 256) {
    const float e = __expf(s_scores[j] - m);
    s_scores[j] = e;
    lsum += e;
  }
  s_red[tid] = lsum;
  __syncthreads();
  for (int s = 128; s > 0; s >>= 1) {
    if (tid < s) s_red[tid] += s_red[tid + s];
    __syncthreads();
  }
  const float inv = 1.0f / s_red[0];
  __syncthreads();

  const int d = tid & (HD_ - 1);
  const int g = tid >> 7;  // 0..1
  const float* vc   = v_cache + ((size_t)b * KVH_ + kvh) * LCACHE_ * HD_;
  const float* vnew = vbuf + (size_t)b * (KVH_ * HD_) + kvh * HD_;
  float acc = 0.f;
  for (int j = g; j <= p; j += 2) {
    const float* vr = (j == p) ? vnew : (vc + (size_t)j * HD_);
    acc += s_scores[j] * vr[d];
  }
  s_part[g][d] = acc;
  __syncthreads();
  if (tid < HD_) {
    const float sum = s_part[0][tid] + s_part[1][tid];
    out[(size_t)b * DMODEL + h * HD_ + tid] = sum * inv;
  }
}

// ---------------------------------------------------------------------------
extern "C" void kernel_launch(void* const* d_in, const int* in_sizes, int n_in,
                              void* d_out, int out_size, void* d_ws, size_t ws_size,
                              hipStream_t stream) {
  (void)in_sizes; (void)n_in; (void)out_size; (void)ws_size;
  const float* x_t        = (const float*)d_in[0];
  const float* cos_t      = (const float*)d_in[1];
  const float* sin_t      = (const float*)d_in[2];
  const float* k_cache    = (const float*)d_in[3];
  const float* v_cache    = (const float*)d_in[4];
  const int*   pos        = (const int*)d_in[5];
  const float* norm_w     = (const float*)d_in[6];
  const float* q_w        = (const float*)d_in[7];
  const float* k_w        = (const float*)d_in[8];
  const float* v_w        = (const float*)d_in[9];
  const float* o_w        = (const float*)d_in[10];
  const float* ffn_norm_w = (const float*)d_in[11];
  const float* gate_w     = (const float*)d_in[12];
  const float* gate_b     = (const float*)d_in[13];
  const float* up_w       = (const float*)d_in[14];
  const float* up_b       = (const float*)d_in[15];
  const float* down_w     = (const float*)d_in[16];
  const float* down_b     = (const float*)d_in[17];
  float* out = (float*)d_out;

  // Workspace layout (floats): total 622,592 floats = ~2.4 MB
  float* ws   = (float*)d_ws;
  float* h    = ws;             // 32*2048
  float* qbuf = h    + 65536;   // 32*2048
  float* kbuf = qbuf + 65536;   // 32*512
  float* vbuf = kbuf + 16384;   // 32*512
  float* attn = vbuf + 16384;   // 32*2048
  float* h2   = attn + 65536;   // 32*2048
  float* f    = h2   + 65536;   // 32*2048
  float* act  = f    + 65536;   // 32*8192

  // 1. input RMSNorm
  rmsnorm_kernel<<<B_, 256, 0, stream>>>(h, x_t, norm_w);
  // 2-4. Q/K/V projections (WMMA bf16, uniform TDM pipeline)
  wmma_gemm_kernel<false, false><<<dim3(DMODEL / 64, 2), 128, 0, stream>>>(
      qbuf, h, q_w, nullptr, nullptr, DMODEL, DMODEL);
  wmma_gemm_kernel<false, false><<<dim3((KVH_ * HD_) / 64, 2), 128, 0, stream>>>(
      kbuf, h, k_w, nullptr, nullptr, KVH_ * HD_, DMODEL);
  wmma_gemm_kernel<false, false><<<dim3((KVH_ * HD_) / 64, 2), 128, 0, stream>>>(
      vbuf, h, v_w, nullptr, nullptr, KVH_ * HD_, DMODEL);
  // 5. RoPE + q/k head RMSNorm
  rope_norm_kernel<<<B_ * (H_ + KVH_), HD_, 0, stream>>>(qbuf, kbuf, cos_t, sin_t);
  // 6. attention over KV cache (fresh row substituted at j == pos)
  attn_kernel<<<dim3(H_, B_), 256, 0, stream>>>(attn, qbuf, kbuf, vbuf, k_cache, v_cache, pos);
  // 7. output projection + residual
  wmma_gemm_kernel<false, true><<<dim3(DMODEL / 64, 2), 128, 0, stream>>>(
      h2, attn, o_w, nullptr, x_t, DMODEL, DMODEL);
  // 8. FFN RMSNorm
  rmsnorm_kernel<<<B_, 256, 0, stream>>>(f, h2, ffn_norm_w);
  // 9. fused gate/up + SiLU
  wmma_gateup_kernel<<<dim3(DFF_ / 64, 2), 128, 0, stream>>>(act, f, gate_w, gate_b, up_w, up_b);
  // 10. down projection + bias + residual -> out
  wmma_gemm_kernel<true, true><<<dim3(DMODEL / 64, 2), 128, 0, stream>>>(
      out, act, down_w, down_b, h2, DMODEL, DFF_);
}
